// ScaledDotProductAttention_14748917694836
// MI455X (gfx1250) — matrix operand order repaired and checked by reading, not compiled, not run on hardware
//
#include <hip/hip_runtime.h>

typedef __attribute__((ext_vector_type(16))) _Float16 v16h;
typedef __attribute__((ext_vector_type(8)))  _Float16 v8h;
typedef __attribute__((ext_vector_type(8)))  float    v8f;
typedef __attribute__((ext_vector_type(4)))  float    v4f;

#define S_LEN 1024
#define D_DIM 64
#define NW 4              // waves per block
#define TPB (NW * 32)

__global__ __launch_bounds__(TPB)
void sdpa_fwd_kernel(const float* __restrict__ Q,
                     const float* __restrict__ K,
                     const float* __restrict__ V,
                     const int*   __restrict__ Msk,
                     float* __restrict__ Out,    // [B,H,S,D]
                     float* __restrict__ Attn,   // [B,H,S,S]
                     int Hn)
{
    __shared__ alignas(16) _Float16 sbuf[16 * S_LEN];      // scores -> unnormalized probs (fp16)
    __shared__ alignas(16) float    obuf[NW][16 * D_DIM];  // per-wave partial P*V
    __shared__ float    red[16 * 8];
    __shared__ float    mrow[16];
    __shared__ float    linv[16];

    const int t    = threadIdx.x;
    const int w    = t >> 5;          // wave id in block
    const int lane = t & 31;
    const int n    = lane & 15;       // lane%16: C's N, A's M, B's N
    const int hi   = lane >> 4;       // half-wave group

    const int qt = blockIdx.x & 63;   // S/16 = 64 query tiles
    const int bh = blockIdx.x >> 6;   // b*H + h
    const int b  = bh / Hn;
    const int q0 = qt * 16;

    const size_t qkvBase = (size_t)bh * S_LEN * D_DIM;
    const float* Qp = Q + qkvBase;
    const float* Kp = K + qkvBase;
    const float* Vp = V + qkvBase;
    const int*   Mp = Msk + (size_t)b * S_LEN * S_LEN;

    // ---- load A = Q tile (16x64), scale folded in (0.125 exact in f16) ----
    // A 16x32 f16 layout: lane%16 = row M; halves[0..7] = K(8*hi ..), halves[8..15] = K(16+8*hi ..)
    const float* qr = Qp + (size_t)(q0 + n) * D_DIM;
    v16h a0, a1;
#pragma unroll
    for (int i = 0; i < 8; ++i) {
        a0[i]     = (_Float16)(qr[      8 * hi + i] * 0.125f);
        a0[i + 8] = (_Float16)(qr[16 +  8 * hi + i] * 0.125f);
        a1[i]     = (_Float16)(qr[32 +  8 * hi + i] * 0.125f);
        a1[i + 8] = (_Float16)(qr[48 +  8 * hi + i] * 0.125f);
    }

    // ---- phase 1: QK^T scores for this wave's 256-key range, mask, -> LDS fp16 ----
    for (int kt = w * 16; kt < w * 16 + 16; ++kt) {
        const int k0 = kt * 16;
        // B 32x16 f16 layout: lane%16 = col N (key), lanes 0-15 hold K=0..7,16..23, 16-31 hold K=8..15,24..31
        const float* kr = Kp + (size_t)(k0 + n) * D_DIM;
        __builtin_prefetch(kr + 16 * D_DIM, 0, 0);   // next key tile -> global_prefetch_b8
        v16h b0, b1;
#pragma unroll
        for (int i = 0; i < 8; ++i) {
            b0[i]     = (_Float16)kr[      8 * hi + i];
            b0[i + 8] = (_Float16)kr[16 +  8 * hi + i];
            b1[i]     = (_Float16)kr[32 +  8 * hi + i];
            b1[i + 8] = (_Float16)kr[48 +  8 * hi + i];
        }
        v8f c = {};
        c = __builtin_amdgcn_wmma_f32_16x16x32_f16(false, a0, false, b0, (short)0, c, false, false);
        c = __builtin_amdgcn_wmma_f32_16x16x32_f16(false, a1, false, b1, (short)0, c, false, false);
        // C layout: N = lane%16, M = vgpr + 8*hi
#pragma unroll
        for (int i = 0; i < 8; ++i) {
            const int r  = 8 * hi + i;
            const int mk = Mp[(size_t)(q0 + r) * S_LEN + k0 + n];
            const float s = (mk == 0) ? -30000.0f : c[i];
            sbuf[r * S_LEN + k0 + n] = (_Float16)s;
        }
    }
    __syncthreads();

    // ---- phase 2: row max (8 threads per row, 128 cols each, b128 LDS reads) ----
    {
        const int r = t >> 3, seg = t & 7;
        const int c0 = seg * 128;
        float mx = -3.0e38f;
#pragma unroll 4
        for (int cc = 0; cc < 16; ++cc) {
            const v8h x = *(const v8h*)&sbuf[r * S_LEN + c0 + cc * 8];
#pragma unroll
            for (int e = 0; e < 8; ++e) mx = fmaxf(mx, (float)x[e]);
        }
        red[t] = mx;
    }
    __syncthreads();
    if (t < 16) {
        float mx = red[t * 8];
#pragma unroll
        for (int i = 1; i < 8; ++i) mx = fmaxf(mx, red[t * 8 + i]);
        mrow[t] = mx;
    }
    __syncthreads();

    // ---- phase 3: exp(s - m) in place (fp16, b128 LDS r/w) + row sum ----
    {
        const int r = t >> 3, seg = t & 7;
        const float m_r = mrow[r];
        const int c0 = seg * 128;
        float sum = 0.0f;
#pragma unroll 2
        for (int cc = 0; cc < 16; ++cc) {
            v8h* p = (v8h*)&sbuf[r * S_LEN + c0 + cc * 8];
            const v8h x = *p;
            v8h e16;
#pragma unroll
            for (int e = 0; e < 8; ++e) {
                const float ev = __expf((float)x[e] - m_r);
                sum += ev;
                e16[e] = (_Float16)ev;
            }
            *p = e16;
        }
        red[t] = sum;
    }
    __syncthreads();
    if (t < 16) {
        float sum = 0.0f;
#pragma unroll
        for (int i = 0; i < 8; ++i) sum += red[t * 8 + i];
        linv[t] = 1.0f / sum;
    }
    __syncthreads();

    // ---- phase 4: stream normalized attn to HBM (b128 loads, 2x b128 stores/thread) ----
    float* attnTile = Attn + ((size_t)bh * S_LEN + q0) * S_LEN;
#pragma unroll 2
    for (int j = 0; j < (16 * S_LEN) / (TPB * 8); ++j) {
        const int chunk = t + j * TPB;          // 8-element chunk id
        const int base  = chunk * 8;
        const int r = base >> 10, c = base & (S_LEN - 1);
        const float li = linv[r];
        const v8h p = *(const v8h*)&sbuf[r * S_LEN + c];
        v4f lo, hi2;
#pragma unroll
        for (int e = 0; e < 4; ++e) {
            lo[e]  = (float)p[e]     * li;
            hi2[e] = (float)p[e + 4] * li;
        }
        float* dst = attnTile + (size_t)r * S_LEN + c;
        *(v4f*)(dst)     = lo;
        *(v4f*)(dst + 4) = hi2;
    }

    // ---- phase 5: P*V for this wave's 256 keys (A from LDS, 4 N-tiles of D) ----
    v8f o0 = {}, o1 = {}, o2 = {}, o3 = {};
    for (int it = 0; it < 8; ++it) {
        const int kb = w * 256 + it * 32;
        __builtin_prefetch(Vp + (size_t)(kb + 32 + 16 * hi) * D_DIM + n, 0, 0); // next V tile
        v16h ap;
#pragma unroll
        for (int i = 0; i < 8; ++i) {
            ap[i]     = sbuf[n * S_LEN + kb + 16 * hi + i];
            ap[i + 8] = sbuf[n * S_LEN + kb + 16 * hi + 8 + i];
        }
        v16h bv0, bv1, bv2, bv3;
#pragma unroll
        for (int kk = 0; kk < 16; ++kk) {
            const float* vr = Vp + (size_t)(kb + 16 * hi + kk) * D_DIM + n;
            bv0[kk] = (_Float16)vr[0];
            bv1[kk] = (_Float16)vr[16];
            bv2[kk] = (_Float16)vr[32];
            bv3[kk] = (_Float16)vr[48];
        }
        o0 = __builtin_amdgcn_wmma_f32_16x16x32_f16(false, ap, false, bv0, (short)0, o0, false, false);
        o1 = __builtin_amdgcn_wmma_f32_16x16x32_f16(false, ap, false, bv1, (short)0, o1, false, false);
        o2 = __builtin_amdgcn_wmma_f32_16x16x32_f16(false, ap, false, bv2, (short)0, o2, false, false);
        o3 = __builtin_amdgcn_wmma_f32_16x16x32_f16(false, ap, false, bv3, (short)0, o3, false, false);
    }
#pragma unroll
    for (int i = 0; i < 8; ++i) {
        const int r = 8 * hi + i;
        obuf[w][r * D_DIM +      n] = o0[i];
        obuf[w][r * D_DIM + 16 + n] = o1[i];
        obuf[w][r * D_DIM + 32 + n] = o2[i];
        obuf[w][r * D_DIM + 48 + n] = o3[i];
    }
    __syncthreads();

    // ---- phase 6: cross-wave reduce, normalize, write output (b128 stores) ----
    float* outTile = Out + ((size_t)bh * S_LEN + q0) * D_DIM;
#pragma unroll
    for (int j = 0; j < (16 * D_DIM) / (TPB * 4); ++j) {
        const int chunk = t + j * TPB;          // 4-element chunk id
        const int base  = chunk * 4;
        const int r = base >> 6, c = base & (D_DIM - 1);
        const v4f s0 = *(const v4f*)&obuf[0][r * D_DIM + c];
        const v4f s1 = *(const v4f*)&obuf[1][r * D_DIM + c];
        const v4f s2 = *(const v4f*)&obuf[2][r * D_DIM + c];
        const v4f s3 = *(const v4f*)&obuf[3][r * D_DIM + c];
        const float li = linv[r];
        v4f s;
#pragma unroll
        for (int e = 0; e < 4; ++e)
            s[e] = (s0[e] + s1[e] + s2[e] + s3[e]) * li;
        *(v4f*)(outTile + (size_t)r * D_DIM + c) = s;
    }
}

extern "C" void kernel_launch(void* const* d_in, const int* in_sizes, int n_in,
                              void* d_out, int out_size, void* d_ws, size_t ws_size,
                              hipStream_t stream) {
    (void)in_sizes; (void)n_in; (void)out_size; (void)d_ws; (void)ws_size;
    const float* q   = (const float*)d_in[0];
    const float* k   = (const float*)d_in[1];
    const float* v   = (const float*)d_in[2];
    const int*   msk = (const int*)d_in[3];

    const int B = 4, H = 16;
    float* out  = (float*)d_out;                               // [B,H,S,D]
    float* attn = out + (size_t)B * H * S_LEN * D_DIM;         // [B,H,S,S]

    dim3 grid(B * H * (S_LEN / 16));   // 4096 blocks
    sdpa_fwd_kernel<<<grid, TPB, 0, stream>>>(q, k, v, msk, out, attn, H);
}
